// FastGRNN_54528904790389
// MI455X (gfx1250) — matrix operand order repaired and checked by reading, not compiled, not run on hardware
//
#include <hip/hip_runtime.h>

// ---------------------------------------------------------------------------
// FastGRNN on MI455X (gfx1250): fused [x_t | h] @ [W | U]^T recurrence with
// bf16 WMMA (v_wmma_f32_16x16x32_bf16), fp32 state carry, per-step kernels.
// Round 2: branch-free gate epilogue using native trans ops (v_tanh_f32 /
// v_exp_f32 / v_rcp_f32) instead of libm tanhf's divergent code path.
// ---------------------------------------------------------------------------

typedef __attribute__((ext_vector_type(16))) __bf16 v16bf;
typedef __attribute__((ext_vector_type(8)))  __bf16 v8bf;
typedef __attribute__((ext_vector_type(4)))  __bf16 v4bf;
typedef __attribute__((ext_vector_type(8)))  float  v8f;

#define BATCH 128
#define SEQ_T 1024
#define IN_I  512
#define HID_H 1024
#define KTOT  1536              // IN_I + HID_H
#define KSTEP 32
#define NKT   (KTOT / KSTEP)    // 48 WMMAs per output tile

// ws layout: [ Wcat bf16 1024x1536 | hbuf f32 128x1024 ]
#define WCAT_ELEMS ((size_t)HID_H * KTOT)   // 1,572,864 bf16 = 3 MB
#define WCAT_BYTES (WCAT_ELEMS * 2)

// ---- branch-free device math on native gfx1250 trans ops -------------------
__device__ __forceinline__ float fast_rcp(float x) {
#if __has_builtin(__builtin_amdgcn_rcpf)
  return __builtin_amdgcn_rcpf(x);      // v_rcp_f32
#else
  return 1.0f / x;
#endif
}

__device__ __forceinline__ float fast_sigmoid(float a) {
  // saturates to 0/1 at both tails, no branches: v_exp_f32 + v_rcp_f32
  return fast_rcp(1.0f + __expf(-a));
}

__device__ __forceinline__ float fast_tanh(float a) {
#if __has_builtin(__builtin_amdgcn_tanhf)
  return __builtin_amdgcn_tanhf(a);     // native v_tanh_f32 (CDNA5 trans op)
#else
  // 1 - 2/(e^{2a}+1): exp overflow -> +1, underflow -> -1, never NaN
  const float e2 = __expf(2.0f * a);
  return 1.0f - 2.0f * fast_rcp(e2 + 1.0f);
#endif
}

// ---------------------------------------------------------------------------
// One-time prep: pack W (HxI) and U (HxH) fp32 -> bf16 concat rows [H x 1536],
// and zero-init h0 (stored in d_out for the even-step ping-pong buffer).
// ---------------------------------------------------------------------------
__global__ void fastgrnn_prep(const float* __restrict__ W,
                              const float* __restrict__ U,
                              __bf16* __restrict__ Wcat,
                              float* __restrict__ h0) {
  size_t i = (size_t)blockIdx.x * blockDim.x + threadIdx.x;
  if (i < WCAT_ELEMS) {
    int j = (int)(i / KTOT);
    int k = (int)(i % KTOT);
    float v = (k < IN_I) ? W[(size_t)j * IN_I + k]
                         : U[(size_t)j * HID_H + (k - IN_I)];
    Wcat[i] = (__bf16)v;
  } else {
    size_t r = i - WCAT_ELEMS;
    if (r < (size_t)BATCH * HID_H) h0[r] = 0.0f;
  }
}

// ---------------------------------------------------------------------------
// One recurrence step:
//   pre  = [x_t | h_prev] @ Wcat^T          (bf16 WMMA, fp32 accum)
//   z    = sigmoid(pre + b_z)
//   ht   = tanh(pre + b_h)
//   h    = (zeta*(1-z) + nu)*ht + z*h_prev
// Grid: (16 N-groups, 8 M-tiles), 128 threads = 4 waves; wave -> one 16x16 tile.
// ---------------------------------------------------------------------------
__global__ void __launch_bounds__(128)
fastgrnn_step(const float* __restrict__ x, int t,
              const __bf16* __restrict__ Wcat,
              const float* __restrict__ bz, const float* __restrict__ bh,
              const float* __restrict__ zeta, const float* __restrict__ nu,
              const float* __restrict__ hprev, float* __restrict__ hnext) {
  __shared__ __bf16 smem[16 * KTOT];   // 48 KB A-panel (16 rows x 1536 K)

  const int tid = threadIdx.x;
  const int m0  = blockIdx.y * 16;     // batch-tile base row

  // ---- Stage A = [x_t | h_prev] rows m0..m0+15, fp32 -> bf16, vectorized ----
  // x part: 16 rows x 512 cols (128 float4 per row)
  for (int i = tid; i < 16 * (IN_I / 4); i += 128) {
    int row = i >> 7;
    int c4  = i & 127;
    const float4 v = *(const float4*)(x + ((size_t)(m0 + row) * SEQ_T + t) * IN_I + 4 * c4);
    v4bf o; o[0] = (__bf16)v.x; o[1] = (__bf16)v.y; o[2] = (__bf16)v.z; o[3] = (__bf16)v.w;
    *(v4bf*)(&smem[row * KTOT + 4 * c4]) = o;
  }
  // h part: 16 rows x 1024 cols (256 float4 per row)
  for (int i = tid; i < 16 * (HID_H / 4); i += 128) {
    int row = i >> 8;
    int c4  = i & 255;
    const float4 v = *(const float4*)(hprev + (size_t)(m0 + row) * HID_H + 4 * c4);
    v4bf o; o[0] = (__bf16)v.x; o[1] = (__bf16)v.y; o[2] = (__bf16)v.z; o[3] = (__bf16)v.w;
    *(v4bf*)(&smem[row * KTOT + IN_I + 4 * c4]) = o;
  }
  __syncthreads();

  // ---- WMMA K-loop: one 16x16 output tile per wave ----
  const int lane = tid & 31;
  const int wv   = tid >> 5;                      // wave 0..3
  const int n    = lane & 15;
  const int hi   = lane >> 4;                     // K half-group select
  const int j    = (blockIdx.x * 4 + wv) * 16 + n;  // output hidden column
  const int arow = lane & 15;                     // A-matrix row (M)

  const __bf16* ap = smem + arow * KTOT;
  const __bf16* bp = Wcat + (size_t)j * KTOT + hi * 16;

  v8f acc = {};
  #pragma unroll 4
  for (int kt = 0; kt < NKT; ++kt) {
    const int kb = kt * KSTEP;
    // A 16x32 bf16: lane holds row (lane&15), K contiguous
    //               (lanes 0-15 -> K kb+0..15, lanes 16-31 -> K kb+16..31)
    v16bf a  = *(const v16bf*)(ap + kb + hi * 16);
    // B 32x16 bf16: lane holds column (lane&15), K contiguous, same order as A
    // (lanes 0-15 -> K kb+0..15, lanes 16-31 -> K kb+16..31)
    v16bf b = *(const v16bf*)(bp + kb);
    acc = __builtin_amdgcn_wmma_f32_16x16x32_bf16(
              false, a, false, b, (short)0, acc, false, false);
  }

  // ---- Gate epilogue (C/D layout: VGPR r -> M = r + 8*hi, N = lane&15) ----
  const float bzj = bz[j];
  const float bhj = bh[j];
  const float zt  = zeta[0];
  const float nuv = nu[0];
  #pragma unroll
  for (int r = 0; r < 8; ++r) {
    const int brow = m0 + r + hi * 8;
    const float pre = acc[r];
    const float z   = fast_sigmoid(pre + bzj);
    const float ht  = fast_tanh(pre + bhj);
    const float ho  = hprev[(size_t)brow * HID_H + j];
    hnext[(size_t)brow * HID_H + j] = (zt * (1.0f - z) + nuv) * ht + z * ho;
  }
}

// ---------------------------------------------------------------------------
extern "C" void kernel_launch(void* const* d_in, const int* in_sizes, int n_in,
                              void* d_out, int out_size, void* d_ws, size_t ws_size,
                              hipStream_t stream) {
  (void)in_sizes; (void)n_in; (void)out_size; (void)ws_size;

  const float* x  = (const float*)d_in[0];   // [B, T, I]
  const float* W  = (const float*)d_in[1];   // [H, I]
  const float* U  = (const float*)d_in[2];   // [H, H]
  const float* bz = (const float*)d_in[3];   // [H]
  const float* bh = (const float*)d_in[4];   // [H]
  const float* zt = (const float*)d_in[5];   // [1]
  const float* nu = (const float*)d_in[6];   // [1]

  float*  hout = (float*)d_out;                              // even-step h (and final)
  __bf16* Wcat = (__bf16*)d_ws;                              // 3 MB packed weights
  float*  hbuf = (float*)((char*)d_ws + WCAT_BYTES);         // odd-step h

  // Pack weights to bf16 and zero h0 (in d_out).
  const size_t prepN = WCAT_ELEMS + (size_t)BATCH * HID_H;
  fastgrnn_prep<<<dim3((unsigned)((prepN + 255) / 256)), dim3(256), 0, stream>>>(
      W, U, Wcat, hout);

  // Sequential scan: one kernel per timestep (stream order = step barrier).
  const dim3 grid(16, 8), blk(128);
  for (int t = 0; t < SEQ_T; ++t) {
    const float* hp = (t & 1) ? hbuf : hout;
    float*       hn = (t & 1) ? hout : hbuf;
    fastgrnn_step<<<grid, blk, 0, stream>>>(x, t, Wcat, bz, bh, zt, nu, hp, hn);
  }
}
